// _SparseMoEClassifierLayer_74148315398465
// MI455X (gfx1250) — compile-verified, matrix operand order checked by reading, not run on hardware
//
#include <hip/hip_runtime.h>

// ---------------- problem constants ----------------
#define N_TOK   8192
#define DMODEL  1024
#define HIDDEN  4096
#define NEXP    8
#define CAP     17408        // 2*N_TOK + NEXP*128  (padded token-list capacity)
#define LN_EPS  1e-5f

typedef unsigned short u16;
typedef __bf16 bf16_t;
typedef bf16_t v16bf __attribute__((ext_vector_type(16)));
typedef float  v8f   __attribute__((ext_vector_type(8)));

typedef int v4i32 __attribute__((ext_vector_type(4)));
typedef __attribute__((address_space(1))) v4i32 as1_v4i;
typedef __attribute__((address_space(3))) v4i32 as3_v4i;

union FragU { uint4 q[2]; v16bf v; };

__device__ __forceinline__ v16bf ld_frag(const u16* p0, const u16* p1) {
    FragU f;
    f.q[0] = *reinterpret_cast<const uint4*>(p0);   // 8 bf16, K run 0
    f.q[1] = *reinterpret_cast<const uint4*>(p1);   // 8 bf16, K run 1
    return f.v;
}

// ---- CDNA5 async global->LDS copy (ASYNCcnt-tracked), 16 bytes ----
__device__ __forceinline__ void async_cp16(const u16* g, u16* l) {
#if __has_builtin(__builtin_amdgcn_global_load_async_to_lds_b128)
    // generic global address == AS1 address; low 32 bits of generic LDS address == AS3 offset
    as1_v4i* gp = (as1_v4i*)(unsigned long long)g;
    as3_v4i* lp = (as3_v4i*)(unsigned int)(unsigned long long)l;
    __builtin_amdgcn_global_load_async_to_lds_b128(gp, lp, 0, 0);
#else
    unsigned lds = (unsigned)(unsigned long long)l;
    unsigned long long ga = (unsigned long long)g;
    asm volatile("global_load_async_to_lds_b128 %0, %1, off"
                 :: "v"(lds), "v"(ga) : "memory");
#endif
}

__device__ __forceinline__ void wait_async0() {
#if __has_builtin(__builtin_amdgcn_s_wait_asynccnt)
    __builtin_amdgcn_s_wait_asynccnt(0);
#else
    asm volatile("s_wait_asynccnt 0x0" ::: "memory");
#endif
}

__device__ __forceinline__ u16 f2bf(float f) {
    unsigned int u = __float_as_uint(f);
    unsigned int r = u + 0x7FFFu + ((u >> 16) & 1u);   // round-to-nearest-even
    return (u16)(r >> 16);
}

__device__ __forceinline__ float gelu_tanh(float v) {
    const float c = 0.7978845608028654f;               // sqrt(2/pi)
    float u = c * (v + 0.044715f * v * v * v);
    float t = __expf(2.0f * u);                        // straight-line v_exp_f32 (TRANS)
    float th = 1.0f - 2.0f / (t + 1.0f);               // tanh(u); inf-safe both tails
    return 0.5f * v * (1.0f + th);
}

// ---------------- init: zero accumulators / lists ----------------
__global__ __launch_bounds__(256) void init_kernel(float* mixed, int* tok_ids,
                                                   float* tok_gate, int* cnt, int* cursor) {
    int i = blockIdx.x * 256 + threadIdx.x;
    if (i < N_TOK * DMODEL) mixed[i] = 0.0f;
    if (i < CAP) { tok_ids[i] = 0; tok_gate[i] = 0.0f; }
    if (i < NEXP) { cnt[i] = 0; cursor[i] = 0; }
}

// ---------------- fp32 -> bf16 conversions (weights transposed) ----------------
__global__ __launch_bounds__(256) void convx_kernel(const float* __restrict__ x,
                                                    u16* __restrict__ Xb) {
    int i = blockIdx.x * 256 + threadIdx.x;
    if (i < N_TOK * DMODEL) Xb[i] = f2bf(x[i]);
}

// W1 [E][D][H] -> W1T [E][H][D]
__global__ __launch_bounds__(256) void convw1_kernel(const float* __restrict__ W1,
                                                     u16* __restrict__ W1T) {
    int o = blockIdx.x * 256 + threadIdx.x;
    int e = o >> 22;
    int r = o & 0x3FFFFF;
    int h = r >> 10;
    int d = r & 1023;
    W1T[o] = f2bf(W1[((size_t)e << 22) + ((size_t)d << 12) + h]);
}

// W2 [E][H][D] -> W2T [E][D][H]
__global__ __launch_bounds__(256) void convw2_kernel(const float* __restrict__ W2,
                                                     u16* __restrict__ W2T) {
    int o = blockIdx.x * 256 + threadIdx.x;
    int e = o >> 22;
    int r = o & 0x3FFFFF;
    int d = r >> 12;
    int h = r & 4095;
    W2T[o] = f2bf(W2[((size_t)e << 22) + ((size_t)h << 10) + d]);
}

// ---------------- router: one wave32 per token ----------------
__global__ __launch_bounds__(256) void router_kernel(const float* __restrict__ x,
                                                     const float* __restrict__ Wr,
                                                     float* __restrict__ probs,
                                                     int* __restrict__ topi,
                                                     float* __restrict__ topg,
                                                     int* __restrict__ cnt) {
    int t = blockIdx.x * 8 + (threadIdx.x >> 5);
    int lane = threadIdx.x & 31;
    const float* xr = x + (size_t)t * DMODEL;
    float acc[NEXP];
#pragma unroll
    for (int e = 0; e < NEXP; ++e) acc[e] = 0.0f;
    for (int k = lane; k < DMODEL; k += 32) {
        float xv = xr[k];
        const float* wr = Wr + k * NEXP;
#pragma unroll
        for (int e = 0; e < NEXP; ++e) acc[e] += xv * wr[e];
    }
#pragma unroll
    for (int e = 0; e < NEXP; ++e)
        for (int o = 16; o > 0; o >>= 1) acc[e] += __shfl_xor(acc[e], o, 32);
    if (lane == 0) {
        float mx = acc[0];
#pragma unroll
        for (int e = 1; e < NEXP; ++e) mx = fmaxf(mx, acc[e]);
        float p[NEXP], s = 0.0f;
#pragma unroll
        for (int e = 0; e < NEXP; ++e) { p[e] = expf(acc[e] - mx); s += p[e]; }
        float invs = 1.0f / s;
#pragma unroll
        for (int e = 0; e < NEXP; ++e) probs[t * NEXP + e] = p[e] * invs;
        int i0 = 0;
#pragma unroll
        for (int e = 1; e < NEXP; ++e) if (acc[e] > acc[i0]) i0 = e;
        int i1 = (i0 == 0) ? 1 : 0;
#pragma unroll
        for (int e = 0; e < NEXP; ++e) if (e != i0 && acc[e] > acc[i1]) i1 = e;
        float g0 = 1.0f / (1.0f + expf(acc[i1] - acc[i0]));   // softmax of top-2
        topi[t * 2] = i0; topi[t * 2 + 1] = i1;
        topg[t * 2] = g0; topg[t * 2 + 1] = 1.0f - g0;
        atomicAdd(&cnt[i0], 1);
        atomicAdd(&cnt[i1], 1);
    }
}

// ---------------- padded prefix offsets (tiles of 128 rows) ----------------
__global__ void offsets_kernel(const int* __restrict__ cnt, int* __restrict__ off) {
    if (threadIdx.x == 0 && blockIdx.x == 0) {
        int o = 0;
        for (int e = 0; e < NEXP; ++e) {
            off[e] = o;
            o += ((cnt[e] + 127) >> 7) << 7;
        }
        off[NEXP] = o;
    }
}

// ---------------- build per-expert token lists ----------------
__global__ __launch_bounds__(256) void build_kernel(const int* __restrict__ topi,
                                                    const float* __restrict__ topg,
                                                    const int* __restrict__ off,
                                                    int* __restrict__ cursor,
                                                    int* __restrict__ tok_ids,
                                                    float* __restrict__ tok_gate) {
    int t = blockIdx.x * 256 + threadIdx.x;
    if (t >= N_TOK) return;
    for (int j = 0; j < 2; ++j) {
        int e = topi[t * 2 + j];
        int pos = atomicAdd(&cursor[e], 1);
        int idx = off[e] + pos;
        tok_ids[idx] = t;
        tok_gate[idx] = topg[t * 2 + j];
    }
}

// ---------------- GEMM1: H = gelu(Xg @ W1[e] + b1[e]), bf16 out ----------------
// block = 256 thr = 8 waves (4M x 2N); tile 128x128, K-step 32, async double-buffer
__global__ __launch_bounds__(256) void gemm1_kernel(const u16* __restrict__ Xb,     // [N][D]
                                                    const u16* __restrict__ W1T,    // [E][H][D]
                                                    const float* __restrict__ b1,   // [E][H]
                                                    const int* __restrict__ cnt,
                                                    const int* __restrict__ off,
                                                    const int* __restrict__ tok_ids,
                                                    u16* __restrict__ Hb) {         // [CAP][H]
    int ex = blockIdx.x;
    int e  = ex >> 6;
    int mt = ex & 63;
    if (mt * 128 >= cnt[e]) return;
    int nt  = blockIdx.y;           // 0..31 (HIDDEN/128)
    int seg = off[e];

    __shared__ __align__(16) u16 lA[2][128 * 32];
    __shared__ __align__(16) u16 lB[2][128 * 32];
    __shared__ int sTok[128];

    int tid = threadIdx.x;
    if (tid < 128) sTok[tid] = tok_ids[seg + mt * 128 + tid];
    __syncthreads();

    int lane = tid & 31, w = tid >> 5;
    int wm = w & 3, wn = w >> 2;
    int l15 = lane & 15, kh = lane >> 4;

    v8f acc[2][4];
#pragma unroll
    for (int i = 0; i < 2; ++i)
#pragma unroll
        for (int j = 0; j < 4; ++j)
#pragma unroll
            for (int q = 0; q < 8; ++q) acc[i][j][q] = 0.0f;

    int row2 = tid >> 1, sg = tid & 1;
    const u16* aSrc = Xb + (size_t)sTok[row2] * DMODEL + sg * 16;
    const u16* bSrc = W1T + ((size_t)e * HIDDEN + nt * 128 + row2) * DMODEL + sg * 16;
    int ldsOff = row2 * 32 + sg * 16;

    auto issue = [&](int buf, int k0) {
        async_cp16(aSrc + k0,     &lA[buf][ldsOff]);
        async_cp16(aSrc + k0 + 8, &lA[buf][ldsOff + 8]);
        async_cp16(bSrc + k0,     &lB[buf][ldsOff]);
        async_cp16(bSrc + k0 + 8, &lB[buf][ldsOff + 8]);
    };
    issue(0, 0);

    for (int k0 = 0; k0 < DMODEL; k0 += 32) {
        int buf = (k0 >> 5) & 1;
        wait_async0();
        __syncthreads();
        if (k0 + 32 < DMODEL) {
            issue(buf ^ 1, k0 + 32);
            __builtin_prefetch(aSrc + k0 + 64, 0, 1);
            __builtin_prefetch(bSrc + k0 + 64, 0, 1);
        }

        v16bf af[2], bfr[4];
#pragma unroll
        for (int tm = 0; tm < 2; ++tm) {
            int m = wm * 32 + tm * 16 + l15;
            const u16* p = &lA[buf][m * 32 + kh * 8];
            af[tm] = ld_frag(p, p + 16);
        }
#pragma unroll
        for (int tn = 0; tn < 4; ++tn) {
            int n = wn * 64 + tn * 16 + l15;
            const u16* p = &lB[buf][n * 32 + kh * 8];
            bfr[tn] = ld_frag(p, p + 16);
        }
#pragma unroll
        for (int tm = 0; tm < 2; ++tm)
#pragma unroll
            for (int tn = 0; tn < 4; ++tn)
                acc[tm][tn] = __builtin_amdgcn_wmma_f32_16x16x32_bf16(
                    false, af[tm], false, bfr[tn], (short)0, acc[tm][tn], false, false);
    }

    size_t rowBase = (size_t)(seg + mt * 128);
#pragma unroll
    for (int tm = 0; tm < 2; ++tm)
#pragma unroll
        for (int tn = 0; tn < 4; ++tn) {
            int hcol = nt * 128 + wn * 64 + tn * 16 + l15;
            float bias = b1[e * HIDDEN + hcol];
#pragma unroll
            for (int vr = 0; vr < 8; ++vr) {
                int m = wm * 32 + tm * 16 + vr + 8 * kh;
                float v = acc[tm][tn][vr] + bias;
                Hb[(rowBase + m) * HIDDEN + hcol] = f2bf(gelu_tanh(v));
            }
        }
}

// ---------------- GEMM2: y = H_e @ W2[e] + b2[e]; mixed[tok] += gate*y ----------------
__global__ __launch_bounds__(256) void gemm2_kernel(const u16* __restrict__ Hb,     // [CAP][H]
                                                    const u16* __restrict__ W2T,    // [E][D][H]
                                                    const float* __restrict__ b2,   // [E][D]
                                                    const int* __restrict__ cnt,
                                                    const int* __restrict__ off,
                                                    const int* __restrict__ tok_ids,
                                                    const float* __restrict__ tok_gate,
                                                    float* __restrict__ mixed) {    // [N][D]
    int ex = blockIdx.x;
    int e  = ex >> 6;
    int mt = ex & 63;
    if (mt * 128 >= cnt[e]) return;
    int nt  = blockIdx.y;            // 0..7 (DMODEL/128)
    int seg = off[e];

    __shared__ __align__(16) u16 lA[2][128 * 32];
    __shared__ __align__(16) u16 lB[2][128 * 32];
    __shared__ int   sTok[128];
    __shared__ float sGate[128];

    int tid = threadIdx.x;
    if (tid < 128) {
        sTok[tid]  = tok_ids[seg + mt * 128 + tid];
        sGate[tid] = tok_gate[seg + mt * 128 + tid];
    }
    __syncthreads();

    int lane = tid & 31, w = tid >> 5;
    int wm = w & 3, wn = w >> 2;
    int l15 = lane & 15, kh = lane >> 4;

    v8f acc[2][4];
#pragma unroll
    for (int i = 0; i < 2; ++i)
#pragma unroll
        for (int j = 0; j < 4; ++j)
#pragma unroll
            for (int q = 0; q < 8; ++q) acc[i][j][q] = 0.0f;

    int row2 = tid >> 1, sg = tid & 1;
    const u16* aSrc = Hb + ((size_t)seg + mt * 128 + row2) * HIDDEN + sg * 16;
    const u16* bSrc = W2T + ((size_t)e * DMODEL + nt * 128 + row2) * HIDDEN + sg * 16;
    int ldsOff = row2 * 32 + sg * 16;

    auto issue = [&](int buf, int k0) {
        async_cp16(aSrc + k0,     &lA[buf][ldsOff]);
        async_cp16(aSrc + k0 + 8, &lA[buf][ldsOff + 8]);
        async_cp16(bSrc + k0,     &lB[buf][ldsOff]);
        async_cp16(bSrc + k0 + 8, &lB[buf][ldsOff + 8]);
    };
    issue(0, 0);

    for (int k0 = 0; k0 < HIDDEN; k0 += 32) {
        int buf = (k0 >> 5) & 1;
        wait_async0();
        __syncthreads();
        if (k0 + 32 < HIDDEN) {
            issue(buf ^ 1, k0 + 32);
            __builtin_prefetch(aSrc + k0 + 64, 0, 1);
            __builtin_prefetch(bSrc + k0 + 64, 0, 1);
        }

        v16bf af[2], bfr[4];
#pragma unroll
        for (int tm = 0; tm < 2; ++tm) {
            int m = wm * 32 + tm * 16 + l15;
            const u16* p = &lA[buf][m * 32 + kh * 8];
            af[tm] = ld_frag(p, p + 16);
        }
#pragma unroll
        for (int tn = 0; tn < 4; ++tn) {
            int n = wn * 64 + tn * 16 + l15;
            const u16* p = &lB[buf][n * 32 + kh * 8];
            bfr[tn] = ld_frag(p, p + 16);
        }
#pragma unroll
        for (int tm = 0; tm < 2; ++tm)
#pragma unroll
            for (int tn = 0; tn < 4; ++tn)
                acc[tm][tn] = __builtin_amdgcn_wmma_f32_16x16x32_bf16(
                    false, af[tm], false, bfr[tn], (short)0, acc[tm][tn], false, false);
    }

#pragma unroll
    for (int tm = 0; tm < 2; ++tm)
#pragma unroll
        for (int tn = 0; tn < 4; ++tn) {
            int dcol = nt * 128 + wn * 64 + tn * 16 + l15;
            float bias = b2[e * DMODEL + dcol];
#pragma unroll
            for (int vr = 0; vr < 8; ++vr) {
                int m = wm * 32 + tm * 16 + vr + 8 * kh;
                int tok = sTok[m];
                float g = sGate[m];                       // 0 for pad rows -> exact +-0 add
                atomicAdd(&mixed[(size_t)tok * DMODEL + dcol], g * (acc[tm][tn][vr] + bias));
            }
        }
}

// ---------------- aux loss: deterministic single-block reduction ----------------
__global__ __launch_bounds__(256) void aux_kernel(const float* __restrict__ probs,
                                                  const int* __restrict__ cnt,
                                                  float* __restrict__ out_aux) {
    __shared__ float sd[NEXP][256];
    int tid = threadIdx.x;
    float p[NEXP];
#pragma unroll
    for (int e = 0; e < NEXP; ++e) p[e] = 0.0f;
    for (int t = tid; t < N_TOK; t += 256) {
        const float* pr = probs + t * NEXP;
#pragma unroll
        for (int e = 0; e < NEXP; ++e) p[e] += pr[e];
    }
#pragma unroll
    for (int e = 0; e < NEXP; ++e) sd[e][tid] = p[e];
    __syncthreads();
    for (int o = 128; o > 0; o >>= 1) {
        if (tid < o)
#pragma unroll
            for (int e = 0; e < NEXP; ++e) sd[e][tid] += sd[e][tid + o];
        __syncthreads();
    }
    if (tid == 0) {
        float aux = 0.0f;
        for (int e = 0; e < NEXP; ++e)
            aux += (sd[e][0] / (float)N_TOK) * ((float)cnt[e] / (float)N_TOK);
        out_aux[0] = (float)NEXP * aux;
    }
}

// ---------------- residual + LayerNorm ----------------
__global__ __launch_bounds__(256) void ln_kernel(const float* __restrict__ x,
                                                 const float* __restrict__ mixed,
                                                 const float* __restrict__ lnw,
                                                 const float* __restrict__ lnb,
                                                 float* __restrict__ out) {
    int t = blockIdx.x, tid = threadIdx.x;
    __shared__ float s1[256], s2[256];
    float r[4], lsum = 0.0f, lsq = 0.0f;
#pragma unroll
    for (int i = 0; i < 4; ++i) {
        int d = tid + i * 256;
        float v = x[(size_t)t * DMODEL + d] + mixed[(size_t)t * DMODEL + d];
        r[i] = v; lsum += v; lsq += v * v;
    }
    s1[tid] = lsum; s2[tid] = lsq;
    __syncthreads();
    for (int o = 128; o > 0; o >>= 1) {
        if (tid < o) { s1[tid] += s1[tid + o]; s2[tid] += s2[tid + o]; }
        __syncthreads();
    }
    float mu  = s1[0] * (1.0f / DMODEL);
    float var = s2[0] * (1.0f / DMODEL) - mu * mu;
    float inv = rsqrtf(var + LN_EPS);
#pragma unroll
    for (int i = 0; i < 4; ++i) {
        int d = tid + i * 256;
        out[(size_t)t * DMODEL + d] = (r[i] - mu) * inv * lnw[d] + lnb[d];
    }
}

// ---------------- launch ----------------
extern "C" void kernel_launch(void* const* d_in, const int* in_sizes, int n_in,
                              void* d_out, int out_size, void* d_ws, size_t ws_size,
                              hipStream_t stream) {
    const float* x   = (const float*)d_in[0];
    const float* Wr  = (const float*)d_in[1];
    const float* W1  = (const float*)d_in[2];
    const float* b1  = (const float*)d_in[3];
    const float* W2  = (const float*)d_in[4];
    const float* b2  = (const float*)d_in[5];
    const float* lnw = (const float*)d_in[6];
    const float* lnb = (const float*)d_in[7];
    float* out = (float*)d_out;

    char* ws = (char*)d_ws;
    size_t o = 0;
    auto take = [&](size_t bytes) { size_t r = o; o = (o + bytes + 255) & ~(size_t)255; return r; };
    u16*   Xb       = (u16*)  (ws + take((size_t)N_TOK * DMODEL * 2));
    u16*   W1T      = (u16*)  (ws + take((size_t)NEXP * HIDDEN * DMODEL * 2));
    u16*   W2T      = (u16*)  (ws + take((size_t)NEXP * DMODEL * HIDDEN * 2));
    u16*   Hb       = (u16*)  (ws + take((size_t)CAP * HIDDEN * 2));
    float* mixed    = (float*)(ws + take((size_t)N_TOK * DMODEL * 4));
    float* probs    = (float*)(ws + take((size_t)N_TOK * NEXP * 4));
    int*   topi     = (int*)  (ws + take((size_t)N_TOK * 2 * 4));
    float* topg     = (float*)(ws + take((size_t)N_TOK * 2 * 4));
    int*   tok_ids  = (int*)  (ws + take((size_t)CAP * 4));
    float* tok_gate = (float*)(ws + take((size_t)CAP * 4));
    int*   cnt      = (int*)  (ws + take(NEXP * 4));
    int*   cursor   = (int*)  (ws + take(NEXP * 4));
    int*   offp     = (int*)  (ws + take((NEXP + 1) * 4));

    init_kernel<<<(N_TOK * DMODEL) / 256, 256, 0, stream>>>(mixed, tok_ids, tok_gate, cnt, cursor);
    convx_kernel<<<(N_TOK * DMODEL) / 256, 256, 0, stream>>>(x, Xb);
    convw1_kernel<<<(NEXP * DMODEL * HIDDEN) / 256, 256, 0, stream>>>(W1, W1T);
    convw2_kernel<<<(NEXP * DMODEL * HIDDEN) / 256, 256, 0, stream>>>(W2, W2T);
    router_kernel<<<N_TOK / 8, 256, 0, stream>>>(x, Wr, probs, topi, topg, cnt);
    offsets_kernel<<<1, 1, 0, stream>>>(cnt, offp);
    build_kernel<<<N_TOK / 256, 256, 0, stream>>>(topi, topg, offp, cursor, tok_ids, tok_gate);
    gemm1_kernel<<<dim3(NEXP * 64, HIDDEN / 128), 256, 0, stream>>>(Xb, W1T, b1, cnt, offp, tok_ids, Hb);
    gemm2_kernel<<<dim3(NEXP * 64, DMODEL / 128), 256, 0, stream>>>(Hb, W2T, b2, cnt, offp, tok_ids, tok_gate, mixed);
    aux_kernel<<<1, 256, 0, stream>>>(probs, cnt, out + (size_t)N_TOK * DMODEL);
    ln_kernel<<<N_TOK, 256, 0, stream>>>(x, mixed, lnw, lnb, out);
}
